// DCNv2_1726576856680
// MI455X (gfx1250) — matrix operand order repaired and checked by reading, not compiled, not run on hardware
//
#include <hip/hip_runtime.h>
#include <hip/hip_bf16.h>

typedef __attribute__((ext_vector_type(16))) _Float16 v16h;
typedef __attribute__((ext_vector_type(8)))  _Float16 v8h;
typedef __attribute__((ext_vector_type(8)))  float    v8f;
typedef __attribute__((ext_vector_type(4)))  float    v4f;

#define IN_C  128
#define OUT_C 128
#define Hh    64
#define Ww    64
#define KK    9
#define KDIM  1152   // 9 * 128
#define HW    4096   // 64*64
#define OCH2  32     // offset-conv out channels padded 27 -> 32

__device__ __forceinline__ int clampi(int v, int lo, int hi) {
  return v < lo ? lo : (v > hi ? hi : v);
}

// ---------------------------------------------------------------------------
// Kernel 0a: NCHW -> NHWC transpose of x into workspace (one-time, 8.4 MB).
// ---------------------------------------------------------------------------
__global__ void __launch_bounds__(256)
xpose_kernel(const float* __restrict__ x, float* __restrict__ xt) {
  int i = blockIdx.x * 256 + threadIdx.x;   // i = ((n*4096 + yx)*128 + c)
  if (i >= 4 * HW * IN_C) return;
  int c  = i & (IN_C - 1);
  int yx = (i >> 7) & (HW - 1);
  int n  = i >> 19;
  xt[i] = x[((size_t)n * IN_C + c) * HW + yx];
}

// ---------------------------------------------------------------------------
// Kernel 0b: weights to f16, K = kk*128 + c ordering.
// ---------------------------------------------------------------------------
__global__ void __launch_bounds__(256)
wcvt_kernel(const float* __restrict__ dcn_w, const float* __restrict__ offw,
            _Float16* __restrict__ bt, _Float16* __restrict__ bt2) {
  int i = blockIdx.x * 256 + threadIdx.x;
  if (i >= (OUT_C + OCH2) * KDIM) return;
  int row = i / KDIM;
  int r   = i % KDIM;
  int kk  = r / IN_C;
  int c   = r % IN_C;
  if (row < OUT_C) {
    bt[i] = (_Float16)dcn_w[(size_t)row * KDIM + c * KK + kk];
  } else {
    int ch = row - OUT_C;
    float v = (ch < 27) ? offw[(size_t)ch * KDIM + c * KK + kk] : 0.0f;
    bt2[(size_t)ch * KDIM + r] = (_Float16)v;
  }
}

// --- gather helpers: fill one 64-wide K-chunk (two 32x32 subtiles) -----------
__device__ __forceinline__ void gather_patch(
    const float* __restrict__ xn, _Float16 (*dst)[32][40],
    int pix, int cc4, int oh, int ow0, int kk, int c0) {
  const int ky = kk / 3, kx = kk % 3;
  const int ih = oh - 1 + ky;
  const int iw = ow0 + pix - 1 + kx;
  const bool vin = ((unsigned)ih < (unsigned)Hh) & ((unsigned)iw < (unsigned)Ww);
  const int off  = clampi(ih, 0, Hh - 1) * Ww + clampi(iw, 0, Ww - 1);
  const float* xrow = xn + (size_t)off * IN_C;
  v4f xa = *(const v4f*)(xrow + c0 + cc4);
  v4f xb = *(const v4f*)(xrow + c0 + 32 + cc4);
#pragma unroll
  for (int j = 0; j < 4; ++j) {
    dst[0][pix][cc4 + j] = (_Float16)(vin ? xa[j] : 0.0f);
    dst[1][pix][cc4 + j] = (_Float16)(vin ? xb[j] : 0.0f);
  }
}

__device__ __forceinline__ void gather_bilin(
    const float* __restrict__ xn, _Float16 (*dst)[32][40],
    int pix, int cc4, int c0,
    int y1i, int x1i, float lh, float lw, float mk) {
  float hh = 1.0f - lh, hw = 1.0f - lw;
  int iy1 = y1i - 1, ix1 = x1i - 1;   // padded coord -> x coord
  int iy2 = iy1 + 1, ix2 = ix1 + 1;
  bool vy1 = (unsigned)iy1 < (unsigned)Hh;
  bool vy2 = (unsigned)iy2 < (unsigned)Hh;
  bool vx1 = (unsigned)ix1 < (unsigned)Ww;
  bool vx2 = (unsigned)ix2 < (unsigned)Ww;
  float w11 = (vy1 && vx1) ? hh * hw * mk : 0.0f;
  float w12 = (vy1 && vx2) ? hh * lw * mk : 0.0f;
  float w21 = (vy2 && vx1) ? lh * hw * mk : 0.0f;
  float w22 = (vy2 && vx2) ? lh * lw * mk : 0.0f;
  int cy1 = clampi(iy1, 0, Hh - 1), cy2 = clampi(iy2, 0, Hh - 1);
  int cx1 = clampi(ix1, 0, Ww - 1), cx2 = clampi(ix2, 0, Ww - 1);
  const float* x11 = xn + (size_t)(cy1 * Ww + cx1) * IN_C;
  const float* x12 = xn + (size_t)(cy1 * Ww + cx2) * IN_C;
  const float* x21 = xn + (size_t)(cy2 * Ww + cx1) * IN_C;
  const float* x22 = xn + (size_t)(cy2 * Ww + cx2) * IN_C;
  int ca = c0 + cc4;
  v4f vA = w11 * *(const v4f*)(x11 + ca) + w12 * *(const v4f*)(x12 + ca)
         + w21 * *(const v4f*)(x21 + ca) + w22 * *(const v4f*)(x22 + ca);
  v4f vB = w11 * *(const v4f*)(x11 + ca + 32) + w12 * *(const v4f*)(x12 + ca + 32)
         + w21 * *(const v4f*)(x21 + ca + 32) + w22 * *(const v4f*)(x22 + ca + 32);
#pragma unroll
  for (int j = 0; j < 4; ++j) {
    dst[0][pix][cc4 + j] = (_Float16)vA[j];
    dst[1][pix][cc4 + j] = (_Float16)vB[j];
  }
}

__device__ __forceinline__ v16h mk_frag(v8h lo, v8h hi) {
  v16h f;
#pragma unroll
  for (int i = 0; i < 8; ++i) { f[i] = lo[i]; f[8 + i] = hi[i]; }
  return f;
}

// ---------------------------------------------------------------------------
// Kernel 1: fused DCNv2. Workgroup = 32 pixels, 8 waves, ping-ponged LDS tiles.
// ---------------------------------------------------------------------------
__global__ void __launch_bounds__(256)
dcn_kernel(const float* __restrict__ xt,
           const float* __restrict__ offb,
           const _Float16* __restrict__ bt,
           const _Float16* __restrict__ bt2,
           float* __restrict__ out) {
  __shared__ float    s_om[32][27];
  __shared__ int      s_y1[32][KK];
  __shared__ int      s_x1[32][KK];
  __shared__ float    s_lh[32][KK];
  __shared__ float    s_lw[32][KK];
  __shared__ float    s_mk[32][KK];
  __shared__ _Float16 s_a[2][2][32][40];  // [buf][subtile][pix][K], 80B rows
  __shared__ float    s_red[4][256];      // phase-A partial reduction

  const int t    = threadIdx.x;
  const int lane = t & 31;
  const int wv   = t >> 5;

  const int p0  = blockIdx.x * 32;       // 32 consecutive pixels: same n, oh
  const int n   = p0 / HW;
  const int rem = p0 % HW;
  const int oh  = rem / Ww;
  const int ow0 = rem % Ww;

  const float* xn = xt + (size_t)n * HW * IN_C;   // NHWC plane for this n

  const int lm  = lane & 15;
  const int lhi = lane >> 4;
  const int pix = lane;     // fill mapping: lane = pixel
  const int cc4 = wv * 4;   // fill mapping: wave = 4-channel group

  // ======================= Phase A: offset conv via WMMA ====================
  {
    v8f acc = {};
    const int tileA = wv & 3;            // output tile 0..3
    const int halfA = wv >> 2;           // which 32-subtile of the 64-chunk
    const int mtA   = tileA & 1;
    const int n0A   = (tileA >> 1) * 16;

    gather_patch(xn, s_a[0], pix, cc4, oh, ow0, /*kk=*/0, /*c0=*/0);
    __syncthreads();

    for (int q = 0; q < 18; ++q) {       // chunk q: kk = q>>1, c0 = (q&1)*64
      const int buf = q & 1;
      const int kk  = q >> 1;
      const int c0  = (q & 1) * 64;

      // fragments for current chunk (ds_load issued before next-chunk fill)
      const v8h* arow = (const v8h*)&s_a[buf][halfA][mtA * 16 + lm][0];
      v16h afrag = mk_frag(arow[lhi], arow[2 + lhi]);
      int kbase = kk * IN_C + c0 + halfA * 32;
      const v8h* br = (const v8h*)(bt2 + (size_t)(n0A + lm) * KDIM + kbase);
      v16h bfrag = mk_frag(br[lhi], br[2 + lhi]);

      if (q < 17) {                      // prefetch+stage chunk q+1
        int q2 = q + 1;
        gather_patch(xn, s_a[buf ^ 1], pix, cc4, oh, ow0, q2 >> 1, (q2 & 1) * 64);
      }
      __syncthreads();

      acc = __builtin_amdgcn_wmma_f32_16x16x32_f16(
          false, afrag, false, bfrag, (short)0, acc, false, false);
    }

    // reduce the two subtile partials per output tile
    if (halfA == 1) {
#pragma unroll
      for (int r = 0; r < 8; ++r) s_red[tileA][r * 32 + lane] = acc[r];
    }
    __syncthreads();
    if (halfA == 0) {
      int ch = n0A + lm;
      if (ch < 27) {
        float b = offb[ch];
#pragma unroll
        for (int r = 0; r < 8; ++r) {
          int pr = mtA * 16 + r + 8 * lhi;
          s_om[pr][ch] = acc[r] + s_red[tileA][r * 32 + lane] + b;
        }
      }
    }
  }
  __syncthreads();

  // ============ Phase B: per-pixel per-tap sample coordinates + mask ========
  for (int item = t; item < 32 * KK; item += 256) {
    int pp = item / KK;
    int kk = item % KK;
    int ky = kk / 3, kx = kk % 3;
    float offy = s_om[pp][2 * kk];
    float offx = s_om[pp][2 * kk + 1];
    float m    = 1.0f / (1.0f + expf(-s_om[pp][18 + kk]));
    float py = (float)(oh + 1)       + (float)(ky - 1) + offy;
    float px = (float)(ow0 + pp + 1) + (float)(kx - 1) + offx;
    py = fminf(fmaxf(py, 0.0f), 65.0f);   // H + 2*PAD - 1
    px = fminf(fmaxf(px, 0.0f), 65.0f);
    float y1 = floorf(py), x1 = floorf(px);
    s_y1[pp][kk] = (int)y1;
    s_x1[pp][kk] = (int)x1;
    s_lh[pp][kk] = py - y1;
    s_lw[pp][kk] = px - x1;
    s_mk[pp][kk] = m;
  }
  __syncthreads();

  // ================= Phase C: main deformable GEMM via WMMA =================
  const int mtile = wv & 1;
  const int nA    = (wv >> 1) * 16;
  const int nB    = nA + 64;

  v8f acc0 = {};
  v8f acc1 = {};

  gather_bilin(xn, s_a[0], pix, cc4, /*c0=*/0,
               s_y1[pix][0], s_x1[pix][0], s_lh[pix][0], s_lw[pix][0], s_mk[pix][0]);
  __syncthreads();

  for (int q = 0; q < 18; ++q) {         // chunk q: kk = q>>1, c0 = (q&1)*64
    const int buf = q & 1;
    const int kk  = q >> 1;
    const int c0  = (q & 1) * 64;

    // A fragments for both subtiles of current chunk
    const v8h* ar0 = (const v8h*)&s_a[buf][0][mtile * 16 + lm][0];
    const v8h* ar1 = (const v8h*)&s_a[buf][1][mtile * 16 + lm][0];
    v16h afA = mk_frag(ar0[lhi], ar0[2 + lhi]);
    v16h afB = mk_frag(ar1[lhi], ar1[2 + lhi]);

    // B fragments: both subtiles x both N-tiles (L2-resident rows)
    int kb = kk * IN_C + c0;
    const v8h* brA = (const v8h*)(bt + (size_t)(nA + lm) * KDIM + kb);
    const v8h* brB = (const v8h*)(bt + (size_t)(nB + lm) * KDIM + kb);
    v16h bfA0 = mk_frag(brA[lhi],     brA[2 + lhi]);
    v16h bfA1 = mk_frag(brA[4 + lhi], brA[6 + lhi]);  // +32 halfs
    v16h bfB0 = mk_frag(brB[lhi],     brB[2 + lhi]);
    v16h bfB1 = mk_frag(brB[4 + lhi], brB[6 + lhi]);

    if (q < 17) {                        // stage chunk q+1 into other buffer
      int q2  = q + 1;
      int kk2 = q2 >> 1;
      gather_bilin(xn, s_a[buf ^ 1], pix, cc4, (q2 & 1) * 64,
                   s_y1[pix][kk2], s_x1[pix][kk2],
                   s_lh[pix][kk2], s_lw[pix][kk2], s_mk[pix][kk2]);
    }
    __syncthreads();

    acc0 = __builtin_amdgcn_wmma_f32_16x16x32_f16(
        false, afA, false, bfA0, (short)0, acc0, false, false);
    acc0 = __builtin_amdgcn_wmma_f32_16x16x32_f16(
        false, afB, false, bfA1, (short)0, acc0, false, false);
    acc1 = __builtin_amdgcn_wmma_f32_16x16x32_f16(
        false, afA, false, bfB0, (short)0, acc1, false, false);
    acc1 = __builtin_amdgcn_wmma_f32_16x16x32_f16(
        false, afB, false, bfB1, (short)0, acc1, false, false);
  }

  // ---- Store: D[r] = out row M = r + 8*lhi (pixel), col N = lm (channel) ----
  float* outn = out + (size_t)n * OUT_C * HW + (size_t)oh * Ww;
#pragma unroll
  for (int r = 0; r < 8; ++r) {
    int pixM = mtile * 16 + r + 8 * lhi;
    int ow   = ow0 + pixM;
    outn[(size_t)(nA + lm) * HW + ow] = acc0[r];
    outn[(size_t)(nB + lm) * HW + ow] = acc1[r];
  }
}

// ---------------------------------------------------------------------------
extern "C" void kernel_launch(void* const* d_in, const int* in_sizes, int n_in,
                              void* d_out, int out_size, void* d_ws, size_t ws_size,
                              hipStream_t stream) {
  (void)in_sizes; (void)n_in; (void)out_size; (void)ws_size;
  const float* x    = (const float*)d_in[0];
  const float* offw = (const float*)d_in[1];
  const float* offb = (const float*)d_in[2];
  const float* dcnw = (const float*)d_in[3];
  float*       out  = (float*)d_out;

  _Float16* bt  = (_Float16*)d_ws;                                     // 294912 B
  _Float16* bt2 = (_Float16*)((char*)d_ws + (size_t)OUT_C * KDIM * 2); // 73728 B
  float*    xtp = (float*)((char*)d_ws + (size_t)(OUT_C + OCH2) * KDIM * 2); // 8 MB

  xpose_kernel<<<(4 * HW * IN_C) / 256, 256, 0, stream>>>(x, xtp);
  wcvt_kernel<<<((OUT_C + OCH2) * KDIM + 255) / 256, 256, 0, stream>>>(dcnw, offw, bt, bt2);
  dcn_kernel<<<(4 * Hh * Ww) / 32, 256, 0, stream>>>(xtp, offb, bt, bt2, out);
}
